// MMINet_76416058130454
// MI455X (gfx1250) — matrix operand order repaired and checked by reading, not compiled, not run on hardware
//
#include <hip/hip_runtime.h>
#include <hip/hip_bf16.h>

typedef __attribute__((ext_vector_type(16))) _Float16 v16h;
typedef __attribute__((ext_vector_type(8)))  float    v8f;

#define BATCH  2048
#define SEQ    2048
#define HIDDEN 30
#define OPSZ   30
#define NINIT  32
#define LOUT   (SEQ - NINIT)   // 2016

// fast sigmoid: v_mul + v_exp + v_add + v_rcp (no IEEE divide expansion)
__device__ __forceinline__ float fsig(float x) {
    return __builtin_amdgcn_rcpf(1.0f + __expf(-x));
}

// ---------------------------------------------------------------------------
// Kernel 1: the full recurrent scan.
// One wave of 32 lanes owns 16 batch rows. lane r and lane r+16 both map to
// row (m0 + (lane&15)); they split the hidden dimension between them.
//
//   WMMA orientation:  D[n, row] = sum_k  h2h_w[n,k] * h2[row,k]
//     A (constant)  = h2h_w, M-dim = n (two 16-row tiles, K padded to 32)
//     B (recurrent) = h2,    N-dim = batch row, K packed per lane
//     D             = lane -> batch row, VGPR -> n   (matches B after xor-16)
// ---------------------------------------------------------------------------
__global__ __launch_bounds__(256) void mmirnn_scan(
    const float* __restrict__ x,      const float* __restrict__ var,
    const float* __restrict__ dnn1_w, const float* __restrict__ dnn1_b,
    const float* __restrict__ x2h_w,  const float* __restrict__ h2h_w,
    const float* __restrict__ dnn2_w, const float* __restrict__ dnn2_b,
    float* __restrict__ pv_out, float* __restrict__ Hws)
{
    const int lane = threadIdx.x & 31;
    const int wave = threadIdx.x >> 5;
    const int hi   = lane >> 4;       // which half of the wave
    const int r    = lane & 15;       // row within the 16-row tile
    const int row  = (blockIdx.x * 8 + wave) * 16 + r;

    // ---- hysteresis per-lane constants: lane owns j = hi*15 + i -----------
    // clip((dB+h1)/r,-1,1)*r == med3(dB+h1, -r, +r)  (r > 0)
    float w1v[15], rj[15], h1[15];
    #pragma unroll
    for (int i = 0; i < 15; ++i) {
        const int j = hi * 15 + i;
        float rv = (float)(j + 1) / (float)(OPSZ + 1);
        rv = rv * rv * rv;
        rj[i]  = rv;
        w1v[i] = dnn1_w[j];
        h1[i]  = var[row * (2 + OPSZ) + 2 + j];    // h1 init = var[:,2:]
    }
    const float va = var[row * (2 + OPSZ) + 0];
    const float vb = var[row * (2 + OPSZ) + 1];
    const float hcon1 = dnn1_b[0] + dnn1_w[30] * va + dnn1_w[31] * vb;
    const float bias2 = dnn2_b[0];

    // ---- D-layout per-register constants (VGPR v -> hidden index n) -------
    // tile0: n0 = v + 8*hi ; tile1: n1 = 16 + v + 8*hi  (zero-pad n >= 30)
    float xwa0[8], xwc0[8], base0[8], w2_0[8], wsum0[8];
    float xwa1[8], xwc1[8], base1[8], w2_1[8], wsum1[8];
    #pragma unroll
    for (int v = 0; v < 8; ++v) {
        const int n0 = v + 8 * hi;
        const int n1 = 16 + v + 8 * hi;
        xwa0[v]  = x2h_w[n0 * 4 + 0];
        xwc0[v]  = x2h_w[n0 * 4 + 1];
        base0[v] = x2h_w[n0 * 4 + 2] * va + x2h_w[n0 * 4 + 3] * vb;
        w2_0[v]  = dnn2_w[n0];
        float s0 = 0.f;
        for (int k = 0; k < HIDDEN; ++k) s0 += h2h_w[n0 * HIDDEN + k];
        wsum0[v] = s0;
        if (n1 < HIDDEN) {
            xwa1[v]  = x2h_w[n1 * 4 + 0];
            xwc1[v]  = x2h_w[n1 * 4 + 1];
            base1[v] = x2h_w[n1 * 4 + 2] * va + x2h_w[n1 * 4 + 3] * vb;
            w2_1[v]  = dnn2_w[n1];
            float s1 = 0.f;
            for (int k = 0; k < HIDDEN; ++k) s1 += h2h_w[n1 * HIDDEN + k];
            wsum1[v] = s1;
        } else {
            xwa1[v] = xwc1[v] = base1[v] = w2_1[v] = wsum1[v] = 0.f;
        }
    }
    float sw2 = 0.f;
    for (int k = 0; k < HIDDEN; ++k) sw2 += dnn2_w[k];
    const float inv_sw2 = 1.0f / sw2;

    // ---- constant A operands: h2h_w in 16-bit A layout ---------------------
    // lane holds W-row (r) / (16+r); element j -> K (contiguous, matches B):
    //   K = 16*hi + j
    v16h a0, a1;
    #pragma unroll
    for (int j = 0; j < 16; ++j) {
        const int K  = 16 * hi + j;
        const int n1 = 16 + r;
        const float w0f = (K < HIDDEN) ? h2h_w[r * HIDDEN + K] : 0.f;
        const float w1f = (n1 < HIDDEN && K < HIDDEN) ? h2h_w[n1 * HIDDEN + K] : 0.f;
        a0[j] = (_Float16)w0f;
        a1[j] = (_Float16)w1f;
    }

    // ---- t = 0 special step (no WMMA: h2 init is uniform per row) ---------
    const size_t xrow = (size_t)row * SEQ * 3;
    float x0 = x[xrow + 0], x1 = x[xrow + 1], x2v = x[xrow + 2];

    float p1 = 0.f;
    #pragma unroll
    for (int i = 0; i < 15; ++i) {
        float u = x1 + h1[i];
        h1[i] = fminf(fmaxf(u, -rj[i]), rj[i]);    // med3 w/ neg modifier
        p1 += h1[i] * w1v[i];
    }
    const float Hh0 = p1 + __shfl_xor(p1, 16, 32) + hcon1;
    const float c   = (x0 - Hh0) * inv_sw2;         // uniform h2 init per row

    float h20[8], h21[8];
    #pragma unroll
    for (int v = 0; v < 8; ++v) {
        h20[v] = fsig(base0[v] + x0 * xwa0[v] + x2v * xwc0[v] + c * wsum0[v]);
        h21[v] = fsig(base1[v] + x0 * xwa1[v] + x2v * xwc1[v] + c * wsum1[v]);
    }

    // ---- pack h2 into the 16-bit B layout (one xor-16 swap per v) ----------
    //  lo lanes: K=0..15  = own h20[k] + partner h20[k-8]
    //  hi lanes: K=16..31 = partner h21[k-16] + own h21[k-24]
    v16h b;
    #pragma unroll
    for (int v = 0; v < 8; ++v) {
        const float send = hi ? h20[v] : h21[v];   // what my partner needs
        const float recv = __shfl_xor(send, 16, 32);
        b[v]     = (_Float16)(hi ? recv   : h20[v]);
        b[v + 8] = (_Float16)(hi ? h21[v] : recv);
    }

    // ---- main recurrence ---------------------------------------------------
    float pv = 0.f, prevH = 0.f, prevBf = 0.f;
    for (int t = 1; t < SEQ; ++t) {
        const size_t xi = (size_t)(row) * SEQ * 3 + (size_t)t * 3;
        const float xt0 = x[xi + 0], xt1 = x[xi + 1], xt2 = x[xi + 2];

        // hysteresis update + dnn1 partial dot
        float pp1 = 0.f;
        #pragma unroll
        for (int i = 0; i < 15; ++i) {
            float u = xt1 + h1[i];
            h1[i] = fminf(fmaxf(u, -rj[i]), rj[i]);
            pp1 += h1[i] * w1v[i];
        }

        // eddy: C = x-input contribution, then WMMA accumulate W x h2
        v8f c0, c1;
        #pragma unroll
        for (int v = 0; v < 8; ++v) {
            c0[v] = base0[v] + xt0 * xwa0[v] + xt2 * xwc0[v];
            c1[v] = base1[v] + xt0 * xwa1[v] + xt2 * xwc1[v];
        }
        c0 = __builtin_amdgcn_wmma_f32_16x16x32_f16(false, a0, false, b,
                                                    (short)0, c0, false, false);
        c1 = __builtin_amdgcn_wmma_f32_16x16x32_f16(false, a1, false, b,
                                                    (short)0, c1, false, false);

        float pp2 = 0.f;
        #pragma unroll
        for (int v = 0; v < 8; ++v) {
            h20[v] = fsig(c0[v]);
            h21[v] = fsig(c1[v]);
            pp2 += h20[v] * w2_0[v] + h21[v] * w2_1[v];
        }
        const float ps = pp1 + pp2;
        const float ot = ps + __shfl_xor(ps, 16, 32) + hcon1 + bias2;

        // repack h2 -> B for next step (8 shuffles)
        #pragma unroll
        for (int v = 0; v < 8; ++v) {
            const float send = hi ? h20[v] : h21[v];
            const float recv = __shfl_xor(send, 16, 32);
            b[v]     = (_Float16)(hi ? recv   : h20[v]);
            b[v + 8] = (_Float16)(hi ? h21[v] : recv);
        }

        // post: trapezoid Pv accumulation + stream scaled H to workspace
        if (t >= NINIT) {
            const float Hs = ot  * 12.6583787f   + 6.99196716e-17f;
            const float Bf = xt0 * 0.0656109348f + (-3.27923689e-19f);
            if (t > NINIT) pv += (Hs + prevH) * (Bf - prevBf);
            prevH = Hs; prevBf = Bf;
            if (!hi) Hws[(size_t)row * LOUT + (t - NINIT)] = Hs;
        }
    }

    if (!hi) {
        const float e = va * 0.268499136f + 5.19875193f;           // log10 scale
        pv_out[row] = 0.5f * pv * __expf(e * 2.302585092994046f);  // 10^e
    }
}

// ---------------------------------------------------------------------------
// Kernel 2: Savitzky-Golay (window 7, poly 2) + roll by N_INIT.
// P[i][j] = (1,i,i^2) . M . (1,j,j^2) / 84,  M = [[64,-39,5],[-39,39,-6],[5,-6,1]]
// ---------------------------------------------------------------------------
__global__ __launch_bounds__(256) void savgol_roll(
    const float* __restrict__ Hws, float* __restrict__ outH)
{
    const int idx = blockIdx.x * blockDim.x + threadIdx.x;
    if (idx >= BATCH * LOUT) return;
    const int rrow = idx / LOUT;
    const int cc   = idx - rrow * LOUT;
    int sc = cc + (LOUT - NINIT);              // real_H = roll(Hs, +NINIT)
    if (sc >= LOUT) sc -= LOUT;

    int ri, start;
    if (sc < 3)               { ri = sc;                      start = 0;        }
    else if (sc >= LOUT - 3)  { ri = 4 + (sc - (LOUT - 3));   start = LOUT - 7; }
    else                      { ri = 3;                       start = sc - 3;   }

    const float rf = (float)ri;
    const float u0 =  64.f - 39.f * rf + 5.f * rf * rf;
    const float u1 = -39.f + 39.f * rf - 6.f * rf * rf;
    const float u2 =   5.f -  6.f * rf +       rf * rf;
    const float inv84 = 1.0f / 84.0f;

    const float* hp = Hws + (size_t)rrow * LOUT + start;
    float acc = 0.f;
    #pragma unroll
    for (int j = 0; j < 7; ++j) {
        const float jf = (float)j;
        const float pj = (u0 + u1 * jf + u2 * jf * jf) * inv84;
        acc += pj * hp[j];
    }
    outH[idx] = acc;
}

// ---------------------------------------------------------------------------
extern "C" void kernel_launch(void* const* d_in, const int* in_sizes, int n_in,
                              void* d_out, int out_size, void* d_ws, size_t ws_size,
                              hipStream_t stream)
{
    const float* x      = (const float*)d_in[0];
    const float* var    = (const float*)d_in[1];
    const float* dnn1_w = (const float*)d_in[2];
    const float* dnn1_b = (const float*)d_in[3];
    const float* x2h_w  = (const float*)d_in[4];
    const float* h2h_w  = (const float*)d_in[5];
    const float* dnn2_w = (const float*)d_in[6];
    const float* dnn2_b = (const float*)d_in[7];

    float* out = (float*)d_out;           // [0,2048): Pv ; [2048,...): real_H
    float* Hws = (float*)d_ws;            // BATCH*LOUT floats of scaled H

    // 16 blocks x 8 waves x 16 rows = 2048 batch rows
    mmirnn_scan<<<BATCH / 128, 256, 0, stream>>>(
        x, var, dnn1_w, dnn1_b, x2h_w, h2h_w, dnn2_w, dnn2_b, out, Hws);

    const int total = BATCH * LOUT;
    savgol_roll<<<(total + 255) / 256, 256, 0, stream>>>(Hws, out + BATCH);
}
